// GraphConvolutionalLayer_40819369181218
// MI455X (gfx1250) — matrix operand order repaired and checked by reading, not compiled, not run on hardware
//
#include <hip/hip_runtime.h>

typedef __attribute__((ext_vector_type(16))) _Float16 v16h;
typedef __attribute__((ext_vector_type(8)))  _Float16 v8h;
typedef __attribute__((ext_vector_type(8)))  float    v8f;
typedef __attribute__((ext_vector_type(4)))  float    v4f;

#define BATCH 32
#define SEQ   1024
#define DIM   128
#define NEG_SLOPE 0.1f

#define WSTR 136   // padded row stride (halves) for W^T/B^T tiles in LDS (16B aligned rows)
#define NSTR 40    // padded row stride (halves) for nodes^T / pooled staging

__device__ __forceinline__ v8f wmma16(v16h a, v16h b, v8f c) {
  // D = A(16x32 f16) x B(32x16 f16) + C(16x16 f32)
  return __builtin_amdgcn_wmma_f32_16x16x32_f16(false, a, false, b, (short)0, c,
                                                false, false);
}

__global__ __launch_bounds__(256)
void gcn_fused(const float* __restrict__ nodes,
               const int*   __restrict__ adj,
               const float* __restrict__ W,
               const float* __restrict__ Bm,
               float* __restrict__ out) {
  __shared__ _Float16 sWtH[DIM * WSTR];
  __shared__ _Float16 sWtL[DIM * WSTR];
  __shared__ _Float16 sBtH[DIM * WSTR];
  __shared__ _Float16 sBtL[DIM * WSTR];
  __shared__ _Float16 sUni[2 * DIM * NSTR];   // nodes^T hi|lo; later pooled hi|lo

  const int tid   = threadIdx.x;
  const int lane  = tid & 31;
  const int wave  = tid >> 5;
  const int m     = lane & 15;         // row (A) / column (B,C) index within tile
  const int hiH   = lane >> 4;         // which half of the wave
  const int b     = blockIdx.x >> 3;
  const int in0   = (blockIdx.x & 7) * 128 + wave * 16;
  const int kbase = hiH ? 16 : 0;      // A-fragment K base for this half
  const int klo   = hiH ? 16 : 0;      // B-fragment K base for this half

  // ---- Phase 0: stage W^T and B^T as hi/lo f16 into LDS (once per block) ----
  for (int i = tid; i < DIM * DIM; i += 256) {
    const int di = i >> 7;     // d_in  (memory row)
    const int dj = i & 127;    // d_out (memory col)
    const float w = W[i];
    const _Float16 wh = (_Float16)w;
    sWtH[dj * WSTR + di] = wh;
    sWtL[dj * WSTR + di] = (_Float16)(w - (float)wh);
    const float x = Bm[i];
    const _Float16 xh = (_Float16)x;
    sBtH[dj * WSTR + di] = xh;
    sBtL[dj * WSTR + di] = (_Float16)(x - (float)xh);
  }

  _Float16* ntH = sUni;                 // nodes^T hi : [DIM][NSTR]
  _Float16* ntL = sUni + DIM * NSTR;    // nodes^T lo

  v8f acc[8] = {};                      // pooled tile: 16 rows x 128 cols
  int icnt = 0;                         // partial in-degree count

  // ---- Phase 1: pooled = mask^T @ nodes, streaming over 'out' (K = 1024) ----
  for (int kk = 0; kk < SEQ / 32; ++kk) {
    const int out0 = kk * 32;
    __syncthreads();
    // stage nodes[b, out0..out0+31, :] transposed + hi/lo split (coalesced b128)
    #pragma unroll
    for (int t = 0; t < 4; ++t) {
      const int i4 = tid + t * 256;                 // float4 index, 0..1023
      const int k  = i4 >> 5;                       // 0..31
      const int d  = (i4 & 31) * 4;
      const v4f v = *(const v4f*)&nodes[((long)(b * SEQ + out0 + k)) * DIM + d];
      #pragma unroll
      for (int e = 0; e < 4; ++e) {
        const _Float16 vh = (_Float16)v[e];
        ntH[(d + e) * NSTR + k] = vh;
        ntL[(d + e) * NSTR + k] = (_Float16)(v[e] - (float)vh);
      }
    }
    __syncthreads();

    // A fragment = mask^T: row (in0+m), K = out0 + kbase + {0..15}
    const int* arow = adj + ((long)(b * SEQ + out0 + kbase)) * SEQ + in0 + m;
    v16h amask;
    #pragma unroll
    for (int j = 0; j < 16; ++j) {
      const int krel = j;                           // 0..15
      const int av = arow[(long)krel * SEQ];
      icnt += (av != 0);
      amask[j] = (av != 0) ? (_Float16)1.0f : (_Float16)0.0f;
    }

    // B fragments (nodes hi/lo) from LDS, 8 column tiles
    #pragma unroll
    for (int nt = 0; nt < 8; ++nt) {
      const v16h bh = *(const v16h*)&ntH[(nt * 16 + m) * NSTR + klo];
      const v16h bl = *(const v16h*)&ntL[(nt * 16 + m) * NSTR + klo];
      acc[nt] = wmma16(amask, bh, acc[nt]);
      acc[nt] = wmma16(amask, bl, acc[nt]);
    }
  }
  __syncthreads();   // all waves done with sUni as nodes^T

  // ---- Phase 2: mean-pool scaling (zero-degree rows -> 0) ----
  const int total = icnt + __shfl_xor(icnt, 16, 32);
  const float scale = (total > 0) ? (1.0f / (float)total) : 0.0f;
  float srow[8];
  #pragma unroll
  for (int r = 0; r < 8; ++r)
    srow[r] = __shfl(scale, (hiH ? 8 : 0) + r, 32);
  #pragma unroll
  for (int nt = 0; nt < 8; ++nt)
    #pragma unroll
    for (int r = 0; r < 8; ++r)
      acc[nt][r] *= srow[r];

  // ---- Phase 3: out = leaky_relu(nodes @ B + pooled @ W), K = 128 in 4 passes
  _Float16* pbH = sUni + wave * (16 * NSTR);                  // wave-private
  _Float16* pbL = sUni + 8 * 16 * NSTR + wave * (16 * NSTR);

  v8f acc2[8] = {};
  #pragma unroll
  for (int pass = 0; pass < 4; ++pass) {
    // stage pooled slice d in [pass*32, pass*32+32), C-layout -> A-layout, hi/lo
    #pragma unroll
    for (int q = 0; q < 2; ++q) {
      const int nt = pass * 2 + q;
      #pragma unroll
      for (int r = 0; r < 8; ++r) {
        const int row  = r + hiH * 8;
        const int dloc = q * 16 + m;
        const float pv = acc[nt][r];
        const _Float16 ph = (_Float16)pv;
        pbH[row * NSTR + dloc] = ph;
        pbL[row * NSTR + dloc] = (_Float16)(pv - (float)ph);
      }
    }
    // A fragments: pooled (from LDS) -- same-wave LDS ops are in-order
    const v8h p0 = *(const v8h*)&pbH[m * NSTR + kbase];
    const v8h p1 = *(const v8h*)&pbH[m * NSTR + kbase + 8];
    const v8h q0 = *(const v8h*)&pbL[m * NSTR + kbase];
    const v8h q1 = *(const v8h*)&pbL[m * NSTR + kbase + 8];
    v16h pah, pal;
    #pragma unroll
    for (int e = 0; e < 8; ++e) {
      pah[e] = p0[e]; pah[e + 8] = p1[e];
      pal[e] = q0[e]; pal[e + 8] = q1[e];
    }
    // A fragments: nodes rows (from global), hi/lo split
    const float* nr = nodes + ((long)(b * SEQ + in0 + m)) * DIM + pass * 32 + kbase;
    const v4f n0 = *(const v4f*)nr;
    const v4f n1 = *(const v4f*)(nr + 4);
    const v4f n2 = *(const v4f*)(nr + 8);
    const v4f n3 = *(const v4f*)(nr + 12);
    v16h nah, nal;
    #pragma unroll
    for (int e = 0; e < 4; ++e) {
      const float a0 = n0[e], a1 = n1[e], a2 = n2[e], a3 = n3[e];
      const _Float16 h0 = (_Float16)a0, h1 = (_Float16)a1;
      const _Float16 h2 = (_Float16)a2, h3 = (_Float16)a3;
      nah[e] = h0; nah[e + 4] = h1; nah[e + 8] = h2; nah[e + 12] = h3;
      nal[e]      = (_Float16)(a0 - (float)h0);
      nal[e + 4]  = (_Float16)(a1 - (float)h1);
      nal[e + 8]  = (_Float16)(a2 - (float)h2);
      nal[e + 12] = (_Float16)(a3 - (float)h3);
    }

    const int kofs = pass * 32 + klo;
    #pragma unroll
    for (int nt2 = 0; nt2 < 8; ++nt2) {
      const int wrow = (nt2 * 16 + m) * WSTR + kofs;
      const v16h wh = *(const v16h*)&sWtH[wrow];
      const v16h wl = *(const v16h*)&sWtL[wrow];
      const v16h bh = *(const v16h*)&sBtH[wrow];
      const v16h bl = *(const v16h*)&sBtL[wrow];
      // (nh+nl)(bh+bl) + (ph+pl)(wh+wl), dropping lo*lo terms
      acc2[nt2] = wmma16(nah, bh, acc2[nt2]);
      acc2[nt2] = wmma16(nal, bh, acc2[nt2]);
      acc2[nt2] = wmma16(nah, bl, acc2[nt2]);
      acc2[nt2] = wmma16(pah, wh, acc2[nt2]);
      acc2[nt2] = wmma16(pal, wh, acc2[nt2]);
      acc2[nt2] = wmma16(pah, wl, acc2[nt2]);
    }
  }

  // ---- Epilogue: leaky ReLU + store ----
  #pragma unroll
  for (int nt = 0; nt < 8; ++nt) {
    #pragma unroll
    for (int r = 0; r < 8; ++r) {
      const float v = acc2[nt][r];
      const float res = fmaxf(v, NEG_SLOPE * v);
      out[((long)(b * SEQ + in0 + r + hiH * 8)) * DIM + nt * 16 + m] = res;
    }
  }
}

extern "C" void kernel_launch(void* const* d_in, const int* in_sizes, int n_in,
                              void* d_out, int out_size, void* d_ws, size_t ws_size,
                              hipStream_t stream) {
  (void)in_sizes; (void)n_in; (void)out_size; (void)d_ws; (void)ws_size;
  const float* nodes = (const float*)d_in[0];
  const int*   adj   = (const int*)d_in[1];
  const float* W     = (const float*)d_in[2];
  const float* Bm    = (const float*)d_in[3];
  float* out = (float*)d_out;
  dim3 grid(BATCH * (SEQ / 128));   // 256 blocks: 32 batches x 8 row tiles
  dim3 block(256);                  // 8 wave32 waves
  hipLaunchKernelGGL(gcn_fused, grid, block, 0, stream, nodes, adj, W, Bm, out);
}
